// SWA_24764781429381
// MI455X (gfx1250) — matrix operand order repaired and checked by reading, not compiled, not run on hardware
//
#include <hip/hip_runtime.h>
#include <hip/hip_bf16.h>

typedef __attribute__((ext_vector_type(16))) _Float16 v16h;
typedef __attribute__((ext_vector_type(8)))  float    v8f;

constexpr int Hc = 192, Wc = 192, Bn = 2, Cn = 64;
constexpr int HWc = Hc * Wc;                 // 36864
constexpr long NCHWc = (long)Bn * Cn * HWc;  // 4718592
constexpr int Hp = 198, Wp = 198;            // padded (halo 3) for async staging

__device__ __forceinline__ int clampi(int v, int lo, int hi) {
  return v < lo ? lo : (v > hi ? hi : v);
}

// ---- WMMA fragment helpers (CDNA5 16-bit layouts, wave32) ------------------
// A 16x32 (MxK): lanes 0-15 M=lane, K {0..7,16..23}; lanes 16-31 same M, K+8.
__device__ __forceinline__ v16h frag_a(const _Float16* a, int lda) {
  int lane = threadIdx.x & 31;
  int m = lane & 15;
  int kh = (lane >> 4) * 8;
  const _Float16* row = a + m * lda;
  v16h f;
#pragma unroll
  for (int v = 0; v < 8; ++v) {
    int k = ((v & 4) << 2) + ((v & 3) << 1) + kh;
    f[2 * v]     = row[k];
    f[2 * v + 1] = row[k + 1];
  }
  return f;
}
// B 32x16 (KxN), stored K-contiguous per column n (ldb = K stride):
// lanes 0-15: N=lane, K {0..7,16..23} ; lanes 16-31: N=lane-16, K+8.
__device__ __forceinline__ v16h frag_b(const _Float16* b, int ldb) {
  int lane = threadIdx.x & 31;
  const _Float16* col = b + (lane & 15) * ldb + (lane >> 4) * 8;
  v16h f;
#pragma unroll
  for (int v = 0; v < 8; ++v) {
    int kk = ((v & 4) << 2) + ((v & 3) << 1);
    f[2 * v]     = col[kk];
    f[2 * v + 1] = col[kk + 1];
  }
  return f;
}
__device__ __forceinline__ v8f wmma16(v16h a, v16h b, v8f c) {
  return __builtin_amdgcn_wmma_f32_16x16x32_f16(false, a, false, b, (short)0, c,
                                                false, false);
}

// Async 16B global->LDS copy (CDNA5, ASYNCcnt-tracked) + wait helper.
__device__ __forceinline__ void async_b128(unsigned lds_byte_addr,
                                           unsigned long long gaddr) {
  asm volatile("global_load_async_to_lds_b128 %0, %1, off"
               :
               : "v"(lds_byte_addr), "v"(gaddr)
               : "memory");
}
__device__ __forceinline__ void wait_async0() {
  asm volatile("s_wait_asynccnt 0" ::: "memory");
}

// ---- weight repack to f16 ---------------------------------------------------
__global__ void k_prep(const float* wf, const float* dwh, const float* dwv,
                       const float* wl, _Float16* wFirstH, _Float16* AHh,
                       _Float16* AHv, _Float16* wLastH) {
  int i = blockIdx.x * 256 + threadIdx.x;
  if (i < 200704) {  // [ky][kx][o][ci] <- w_first[o][ci][ky][kx]
    int ci = i & 63, o = (i >> 6) & 63, t = i >> 12;
    int kx = t % 7, ky = t / 7;
    wFirstH[i] = (_Float16)wf[(o * 64 + ci) * 49 + ky * 7 + kx];
  }
  int j = i - 200704;
  if (j >= 0 && j < 28672) AHh[j] = (_Float16)dwh[j];   // [o][i*7+k]
  j -= 28672;
  if (j >= 0 && j < 28672) AHv[j] = (_Float16)dwv[j];
  j -= 28672;
  if (j >= 0 && j < 4096)  wLastH[j] = (_Float16)wl[j]; // [o][i]
}

// ---- NCHW f32 -> padded NHWC f16 (for async DMA staging) -------------------
__global__ void k_cvtpad(const float* __restrict__ x, _Float16* __restrict__ xHp) {
  long i = (long)blockIdx.x * 256 + threadIdx.x;  // total Bn*Hp*Wp*64
  int ci = (int)(i & 63);
  long t = i >> 6;
  int xp = (int)(t % Wp);
  int yp = (int)((t / Wp) % Hp);
  int b = (int)(t / ((long)Wp * Hp));
  int xu = xp - 3, yu = yp - 3;
  float v = 0.f;
  if (xu >= 0 && xu < Wc && yu >= 0 && yu < Hc)
    v = x[((b * Cn + ci) * Hc + yu) * Wc + xu];
  xHp[i] = (_Float16)v;
}

// ---- 7x7 conv (implicit GEMM, WMMA, async-DMA LDS staging) -----------------
__global__ __launch_bounds__(256) void k_conv7(const _Float16* __restrict__ xHp,
                                               const _Float16* __restrict__ wH,
                                               float* __restrict__ out) {
  __shared__ _Float16 sIn[38 * 64];
  int tid = threadIdx.x;
  int x0 = blockIdx.x * 32, y = blockIdx.y, b = blockIdx.z;
  int wave = tid >> 5, lane = tid & 31;
  int wm = wave >> 1, wn = wave & 1;
  unsigned ldsBase = (unsigned)(unsigned long long)(uintptr_t)(void*)sIn;
  v8f acc = {};
  for (int ky = 0; ky < 7; ++ky) {
    // padded coords: row y+ky, columns [x0, x0+38) -> contiguous 4864B run
    const _Float16* gsrc =
        xHp + (((long)b * Hp + (y + ky)) * Wp + x0) * 64;
    unsigned long long gbase = (unsigned long long)(uintptr_t)gsrc;
    for (int idx = tid; idx < 304; idx += 256)  // 304 x 16B = 4864B
      async_b128(ldsBase + idx * 16, gbase + (unsigned long long)idx * 16);
    wait_async0();
    __syncthreads();
#pragma unroll
    for (int kx = 0; kx < 7; ++kx) {
#pragma unroll
      for (int ks = 0; ks < 2; ++ks) {
        v16h a  = frag_a(wH + (((ky * 7 + kx) * 64 + wm * 16) * 64 + ks * 32), 64);
        v16h bf = frag_b(sIn + ((wn * 16 + kx) * 64 + ks * 32), 64);
        acc = wmma16(a, bf, acc);
      }
    }
    __syncthreads();
  }
  int n = lane & 15, mh = (lane >> 4) * 8;
#pragma unroll
  for (int r = 0; r < 8; ++r) {
    int o = wm * 16 + mh + r;
    out[((b * Cn + o) * Hc + y) * Wc + (x0 + wn * 16 + n)] = acc[r];
  }
}

// ---- separable 7x7 avg pool -------------------------------------------------
__global__ void k_hsum7(const float* __restrict__ in, float* __restrict__ out) {
  long i = (long)blockIdx.x * 256 + threadIdx.x;
  if (i >= NCHWc) return;
  int xp = (int)(i % Wc);
  long base = i - xp;
  float s = 0.f;
#pragma unroll
  for (int j = -3; j <= 3; ++j) {
    int xx = xp + j;
    if (xx >= 0 && xx < Wc) s += in[base + xx];
  }
  out[i] = s;
}
__global__ void k_vsum7(const float* __restrict__ in, float* __restrict__ out) {
  long i = (long)blockIdx.x * 256 + threadIdx.x;
  if (i >= NCHWc) return;
  int xp = (int)(i % Wc);
  int yp = (int)((i / Wc) % Hc);
  long base = i - (long)yp * Wc - xp;
  float s = 0.f;
#pragma unroll
  for (int j = -3; j <= 3; ++j) {
    int yy = yp + j;
    if (yy >= 0 && yy < Hc) s += in[base + (long)yy * Wc + xp];
  }
  out[i] = s * (1.f / 49.f);
}

// ---- fused depthwise 1-D chain (9 -> 7 -> 5 taps) --------------------------
__global__ __launch_bounds__(192) void k_dwchain(
    const float* __restrict__ in, float* __restrict__ out,
    const float* __restrict__ w1, const float* __restrict__ b1,
    const float* __restrict__ w2, const float* __restrict__ b2,
    const float* __restrict__ w3, const float* __restrict__ b3, int vertical) {
  __shared__ float s0[192], s1[192];
  int id = blockIdx.x;
  int line = id % 192;
  int c = (id / 192) % Cn;
  int b = id / (192 * Cn);
  int t = threadIdx.x;
  long base;
  int stride;
  if (vertical) { base = ((long)(b * Cn + c) * Hc) * Wc + line; stride = Wc; }
  else          { base = ((long)(b * Cn + c) * Hc + line) * Wc; stride = 1; }
  s0[t] = in[base + (long)t * stride];
  __syncthreads();
  float acc = b1[c];
#pragma unroll
  for (int j = 0; j < 9; ++j) {
    int u = t - 4 + j;
    if (u >= 0 && u < 192) acc += w1[c * 9 + j] * s0[u];
  }
  s1[t] = acc;
  __syncthreads();
  acc = b2[c];
#pragma unroll
  for (int j = 0; j < 7; ++j) {
    int u = t - 3 + j;
    if (u >= 0 && u < 192) acc += w2[c * 7 + j] * s1[u];
  }
  s0[t] = acc;
  __syncthreads();
  acc = b3[c];
#pragma unroll
  for (int j = 0; j < 5; ++j) {
    int u = t - 2 + j;
    if (u >= 0 && u < 192) acc += w3[c * 5 + j] * s0[u];
  }
  out[base + (long)t * stride] = acc;
}

// ---- offset conv 3x3 (64 -> 14) + BN + tanh --------------------------------
__global__ __launch_bounds__(256) void k_offconv(
    const float* __restrict__ feat, const float* __restrict__ ow,
    const float* __restrict__ ob, const float* __restrict__ bg,
    const float* __restrict__ bb, const float* __restrict__ bm,
    const float* __restrict__ bv, float* __restrict__ off) {
  __shared__ float sF[3 * 18 * 64];
  __shared__ float sW[14 * 576];
  int tid = threadIdx.x;
  int x0 = blockIdx.x * 16, y = blockIdx.y, b = blockIdx.z;
  for (int idx = tid; idx < 3 * 18 * 64; idx += 256) {
    int ci = idx & 63;
    int xl = (idx >> 6) % 18;
    int ky = idx / (18 * 64);
    int yy = y + ky - 1, xx = x0 + xl - 1;
    float v = 0.f;
    if (yy >= 0 && yy < Hc && xx >= 0 && xx < Wc)
      v = feat[((b * Cn + ci) * Hc + yy) * Wc + xx];
    sF[(ky * 18 + xl) * 64 + ci] = v;
  }
  for (int idx = tid; idx < 14 * 576; idx += 256) sW[idx] = ow[idx];
  __syncthreads();
  if (tid < 224) {
    int oc = tid >> 4, pl = tid & 15;
    float acc = ob[oc];
#pragma unroll
    for (int ky = 0; ky < 3; ++ky)
#pragma unroll
      for (int kx = 0; kx < 3; ++kx) {
        const float* fp = &sF[(ky * 18 + pl + kx) * 64];
        const float* wp = &sW[oc * 576 + ky * 3 + kx];  // stride 9 over ci
        for (int ci = 0; ci < 64; ++ci) acc += fp[ci] * wp[ci * 9];
      }
    float inv = bg[oc] * rsqrtf(bv[oc] + 1e-5f);
    float o = acc * inv + (bb[oc] - bm[oc] * inv);
    off[((b * 14 + oc) * Hc + y) * Wc + (x0 + pl)] = tanhf(o);
  }
}

// ---- fused deformable sampling + einsum GEMM (WMMA) ------------------------
__global__ __launch_bounds__(256) void k_dsconv(
    const float* __restrict__ feat, const float* __restrict__ off,
    const _Float16* __restrict__ AH, const float* __restrict__ bias,
    float* __restrict__ out, int morph) {
  __shared__ _Float16 sB[32 * 448];
  __shared__ int sY0[224], sY1[224], sX0[224], sX1[224];
  __shared__ float sWa0[224], sWc0[224], sWa1[224], sWc1[224];
  int tid = threadIdx.x;
  int x0 = blockIdx.x * 32, y = blockIdx.y, b = blockIdx.z;

  if (tid < 32) {  // coordinates for this pixel, all 7 taps
    int px = x0 + tid;
    float o[7];
#pragma unroll
    for (int k = 0; k < 7; ++k) {
      int ch = morph ? (7 + k) : k;
      o[k] = off[((b * 14 + ch) * Hc + y) * Wc + px];
    }
    float cum[7];
    cum[3] = 0.f;
    cum[4] = o[4];
    cum[5] = o[4] + o[5];
    cum[2] = o[2];
    cum[1] = o[2] + o[1];
    cum[0] = o[0];
    cum[6] = o[6];
#pragma unroll
    for (int k = 0; k < 7; ++k) {
      float ys, xs;
      if (morph == 0) { ys = (float)y + cum[k]; xs = (float)(px + k - 3); }
      else            { ys = (float)(y + k - 3); xs = (float)px + cum[k]; }
      float yf = floorf(ys), xf = floorf(xs);
      int it = tid * 7 + k;
      sY0[it] = clampi((int)yf, 0, Hc - 1);
      sY1[it] = clampi((int)yf + 1, 0, Hc - 1);
      sX0[it] = clampi((int)xf, 0, Wc - 1);
      sX1[it] = clampi((int)xf + 1, 0, Wc - 1);
      float y0c = fminf(fmaxf(yf, 0.f), (float)Hc);
      float y1c = fminf(fmaxf(yf + 1.f, 0.f), (float)Hc);
      float x0c = fminf(fmaxf(xf, 0.f), (float)Wc);
      float x1c = fminf(fmaxf(xf + 1.f, 0.f), (float)Wc);
      sWa0[it] = (y1c - ys) * (x1c - xs);
      sWc0[it] = (y1c - ys) * (xs - x0c);
      sWa1[it] = (ys - y0c) * (x1c - xs);
      sWc1[it] = (ys - y0c) * (xs - x0c);
    }
  }
  __syncthreads();
  if (tid < 224) {  // bilinear gather 64 channels -> f16 B tile [px][ci*7+k]
    int pl = tid / 7, k = tid % 7;
    int y0i = sY0[tid], y1i = sY1[tid], x0i = sX0[tid], x1i = sX1[tid];
    float wa0 = sWa0[tid], wc0 = sWc0[tid], wa1 = sWa1[tid], wc1 = sWc1[tid];
    const float* fb = feat + (long)b * Cn * HWc;
    for (int ci = 0; ci < 64; ++ci) {
      const float* fc = fb + (long)ci * HWc;
      float v = wa0 * fc[y0i * Wc + x0i] + wc0 * fc[y0i * Wc + x1i] +
                wa1 * fc[y1i * Wc + x0i] + wc1 * fc[y1i * Wc + x1i];
      sB[pl * 448 + ci * 7 + k] = (_Float16)v;
    }
  }
  __syncthreads();
  int wave = tid >> 5, lane = tid & 31;
  int wm = wave >> 1, wn = wave & 1;
  v8f acc = {};
#pragma unroll
  for (int ch = 0; ch < 14; ++ch) {
    v16h a  = frag_a(AH + (wm * 16) * 448 + ch * 32, 448);
    v16h bf = frag_b(sB + (wn * 16) * 448 + ch * 32, 448);
    acc = wmma16(a, bf, acc);
  }
  int n = lane & 15, mh = (lane >> 4) * 8;
#pragma unroll
  for (int r = 0; r < 8; ++r) {
    int oc = wm * 16 + mh + r;
    out[((b * Cn + oc) * Hc + y) * Wc + (x0 + wn * 16 + n)] = acc[r] + bias[oc];
  }
}

// ---- group-norm statistics (16 groups of 4 channels) -----------------------
__global__ __launch_bounds__(256) void k_gnstats(const float* __restrict__ xin,
                                                 float* __restrict__ stats) {
  __shared__ float rs[256], rq[256];
  int bg = blockIdx.x;  // b*16 + g
  int b = bg >> 4, g = bg & 15;
  const float* base = xin + ((long)(b * Cn + g * 4)) * HWc;
  float s = 0.f, q = 0.f;
  for (int i = threadIdx.x; i < 4 * HWc; i += 256) {
    float v = base[i];
    s += v;
    q += v * v;
  }
  rs[threadIdx.x] = s;
  rq[threadIdx.x] = q;
  __syncthreads();
  for (int o = 128; o > 0; o >>= 1) {
    if (threadIdx.x < o) {
      rs[threadIdx.x] += rs[threadIdx.x + o];
      rq[threadIdx.x] += rq[threadIdx.x + o];
    }
    __syncthreads();
  }
  if (threadIdx.x == 0) {
    float mu = rs[0] * (1.f / (4.f * HWc));
    float var = rq[0] * (1.f / (4.f * HWc)) - mu * mu;
    stats[bg * 2] = mu;
    stats[bg * 2 + 1] = var;
  }
}

// ---- finale: GN+relu(h)+GN+relu(v)+x1 -> 1x1 conv (WMMA) -> BN -> SiLU -----
__global__ __launch_bounds__(256) void k_final(
    const float* __restrict__ dsH, const float* __restrict__ dsV,
    const float* __restrict__ x1, const float* __restrict__ stH,
    const float* __restrict__ stV, const float* __restrict__ ghg,
    const float* __restrict__ ghb, const float* __restrict__ gvg,
    const float* __restrict__ gvb, const _Float16* __restrict__ wLastH,
    const float* __restrict__ lg, const float* __restrict__ lb,
    const float* __restrict__ lm, const float* __restrict__ lv,
    float* __restrict__ out) {
  __shared__ _Float16 sS[32 * 64];
  int tid = threadIdx.x;
  int x0 = blockIdx.x * 32, y = blockIdx.y, b = blockIdx.z;
  long pixBase = (long)y * Wc + x0;
  for (int idx = tid; idx < 32 * 64; idx += 256) {
    int ci = idx & 63, pl = idx >> 6;
    long gi = ((long)(b * Cn + ci)) * HWc + pixBase + pl;
    int g = ci >> 2;
    float mh = stH[(b * 16 + g) * 2], vh = stH[(b * 16 + g) * 2 + 1];
    float hv = (dsH[gi] - mh) * rsqrtf(vh + 1e-5f) * ghg[ci] + ghb[ci];
    hv = fmaxf(hv, 0.f);
    float mv = stV[(b * 16 + g) * 2], vv = stV[(b * 16 + g) * 2 + 1];
    float vl = (dsV[gi] - mv) * rsqrtf(vv + 1e-5f) * gvg[ci] + gvb[ci];
    vl = fmaxf(vl, 0.f);
    sS[pl * 64 + ci] = (_Float16)(hv + vl + x1[gi]);
  }
  __syncthreads();
  int wave = tid >> 5, lane = tid & 31;
  int wm = wave >> 1, wn = wave & 1;
  v8f acc = {};
#pragma unroll
  for (int ch = 0; ch < 2; ++ch) {
    v16h a  = frag_a(wLastH + (wm * 16) * 64 + ch * 32, 64);
    v16h bf = frag_b(sS + (wn * 16) * 64 + ch * 32, 64);
    acc = wmma16(a, bf, acc);
  }
  int n = lane & 15, mh8 = (lane >> 4) * 8;
#pragma unroll
  for (int r = 0; r < 8; ++r) {
    int oc = wm * 16 + mh8 + r;
    float inv = lg[oc] * rsqrtf(lv[oc] + 1e-3f);
    float t = acc[r] * inv + (lb[oc] - lm[oc] * inv);
    out[((b * Cn + oc) * Hc + y) * Wc + (x0 + wn * 16 + n)] = t / (1.f + expf(-t));
  }
}

extern "C" void kernel_launch(void* const* d_in, const int* in_sizes, int n_in,
                              void* d_out, int out_size, void* d_ws,
                              size_t ws_size, hipStream_t stream) {
  (void)in_sizes; (void)n_in; (void)out_size; (void)ws_size;
  const float* x    = (const float*)d_in[0];
  const float* wfir = (const float*)d_in[1];
  const float* hc1w = (const float*)d_in[2];  const float* hc1b = (const float*)d_in[3];
  const float* hc2w = (const float*)d_in[4];  const float* hc2b = (const float*)d_in[5];
  const float* hc3w = (const float*)d_in[6];  const float* hc3b = (const float*)d_in[7];
  const float* vc1w = (const float*)d_in[8];  const float* vc1b = (const float*)d_in[9];
  const float* vc2w = (const float*)d_in[10]; const float* vc2b = (const float*)d_in[11];
  const float* vc3w = (const float*)d_in[12]; const float* vc3b = (const float*)d_in[13];
  const float* h_off_w = (const float*)d_in[14]; const float* h_off_b = (const float*)d_in[15];
  const float* h_bn_g  = (const float*)d_in[16]; const float* h_bn_b  = (const float*)d_in[17];
  const float* h_bn_m  = (const float*)d_in[18]; const float* h_bn_v  = (const float*)d_in[19];
  const float* h_dsc_w = (const float*)d_in[20]; const float* h_dsc_b = (const float*)d_in[21];
  const float* h_gn_g  = (const float*)d_in[22]; const float* h_gn_b  = (const float*)d_in[23];
  const float* v_off_w = (const float*)d_in[24]; const float* v_off_b = (const float*)d_in[25];
  const float* v_bn_g  = (const float*)d_in[26]; const float* v_bn_b  = (const float*)d_in[27];
  const float* v_bn_m  = (const float*)d_in[28]; const float* v_bn_v  = (const float*)d_in[29];
  const float* v_dsc_w = (const float*)d_in[30]; const float* v_dsc_b = (const float*)d_in[31];
  const float* v_gn_g  = (const float*)d_in[32]; const float* v_gn_b  = (const float*)d_in[33];
  const float* last_w  = (const float*)d_in[34];
  const float* last_g  = (const float*)d_in[35]; const float* last_b = (const float*)d_in[36];
  const float* last_m  = (const float*)d_in[37]; const float* last_v = (const float*)d_in[38];

  char* ws = (char*)d_ws;
  size_t cur = 0;
  auto take = [&](size_t bytes) {
    char* p = ws + cur;
    cur += (bytes + 255) & ~(size_t)255;
    return p;
  };
  _Float16* wFirstH = (_Float16*)take(200704 * 2);
  _Float16* AHh     = (_Float16*)take(28672 * 2);
  _Float16* AHv     = (_Float16*)take(28672 * 2);
  _Float16* wLastH  = (_Float16*)take(4096 * 2);
  _Float16* xHp     = (_Float16*)take((size_t)Bn * Hp * Wp * 64 * 2);
  float* t0     = (float*)take((size_t)NCHWc * 4);
  float* t1     = (float*)take((size_t)NCHWc * 4);
  float* x1     = (float*)take((size_t)NCHWc * 4);
  float* dsH    = (float*)take((size_t)NCHWc * 4);
  float* dsV    = (float*)take((size_t)NCHWc * 4);
  float* offBuf = (float*)take((size_t)Bn * 14 * HWc * 4);
  float* stats  = (float*)take(2 * 64 * 4);

  k_prep<<<1024, 256, 0, stream>>>(wfir, h_dsc_w, v_dsc_w, last_w, wFirstH, AHh,
                                   AHv, wLastH);
  {
    long total = (long)Bn * Hp * Wp * 64;  // 5018112 = 19602 * 256
    k_cvtpad<<<(int)(total / 256), 256, 0, stream>>>(x, xHp);
  }
  k_conv7<<<dim3(Wc / 32, Hc, Bn), 256, 0, stream>>>(xHp, wFirstH, t0);
  k_hsum7<<<(int)(NCHWc / 256), 256, 0, stream>>>(t0, t1);
  k_vsum7<<<(int)(NCHWc / 256), 256, 0, stream>>>(t1, x1);
  k_dwchain<<<Bn * Cn * 192, 192, 0, stream>>>(x1, t0, hc1w, hc1b, hc2w, hc2b,
                                               hc3w, hc3b, 0);
  k_dwchain<<<Bn * Cn * 192, 192, 0, stream>>>(x1, t1, vc1w, vc1b, vc2w, vc2b,
                                               vc3w, vc3b, 1);
  // horizontal deformable path
  k_offconv<<<dim3(Wc / 16, Hc, Bn), 256, 0, stream>>>(
      t0, h_off_w, h_off_b, h_bn_g, h_bn_b, h_bn_m, h_bn_v, offBuf);
  k_dsconv<<<dim3(Wc / 32, Hc, Bn), 256, 0, stream>>>(t0, offBuf, AHh, h_dsc_b,
                                                      dsH, 0);
  k_gnstats<<<32, 256, 0, stream>>>(dsH, stats);
  // vertical deformable path
  k_offconv<<<dim3(Wc / 16, Hc, Bn), 256, 0, stream>>>(
      t1, v_off_w, v_off_b, v_bn_g, v_bn_b, v_bn_m, v_bn_v, offBuf);
  k_dsconv<<<dim3(Wc / 32, Hc, Bn), 256, 0, stream>>>(t1, offBuf, AHv, v_dsc_b,
                                                      dsV, 1);
  k_gnstats<<<32, 256, 0, stream>>>(dsV, stats + 64);
  // merge + 1x1 conv + BN + SiLU
  k_final<<<dim3(Wc / 32, Hc, Bn), 256, 0, stream>>>(
      dsH, dsV, x1, stats, stats + 64, h_gn_g, h_gn_b, v_gn_g, v_gn_b, wLastH,
      last_g, last_b, last_m, last_v, (float*)d_out);
}
